// MPBlock_36988258353722
// MI455X (gfx1250) — matrix operand order repaired and checked by reading, not compiled, not run on hardware
//
#include <hip/hip_runtime.h>

typedef __attribute__((ext_vector_type(16))) _Float16 v16h;
typedef __attribute__((ext_vector_type(8)))  float    v8f;

#define DD 256          // feature dim
#define KT 8            // 256 / 32 K-tiles
#define NT 16           // 256 / 16 N-tiles
#define SLICE_H 8192    // halves per kt weight slice: 16 nt * 32 lanes * 16 j
#define SLICE_B 16384   // bytes per slice

// ---------------------------------------------------------------- zero
__global__ void zero_f32(float* __restrict__ p, long long n) {
  long long i = (long long)blockIdx.x * blockDim.x + threadIdx.x;
  long long stride = (long long)gridDim.x * blockDim.x;
  for (; i < n; i += stride) p[i] = 0.0f;
}

// ---------------------------------------------------------------- degree (counted once; all 3 aggregations share dst)
__global__ void deg_count(const int* __restrict__ dst, float* __restrict__ deg, int E) {
  int e = blockIdx.x * blockDim.x + threadIdx.x;
  if (e < E) atomicAdd(&deg[dst[e]], 1.0f);
}

// ---------------------------------------------------------------- scatter-add edge_attr -> sum[dst]
__global__ void scatter_edge(const float* __restrict__ vals, const int* __restrict__ dst,
                             float* __restrict__ sum, int E) {
  int e = blockIdx.x;            // one edge per block (uniform dst load -> scalar path)
  int t = threadIdx.x;           // feature index
  int d = dst[e];
  atomicAdd(&sum[(long long)d * DD + t], vals[(long long)e * DD + t]);
}

// ---------------------------------------------------------------- gather x[src] and scatter-add -> sum[dst]
__global__ void scatter_gather(const float* __restrict__ x, const int* __restrict__ src,
                               const int* __restrict__ dst, float* __restrict__ sum, int E) {
  int e = blockIdx.x;
  int t = threadIdx.x;
  int s = src[e];
  int d = dst[e];
  atomicAdd(&sum[(long long)d * DD + t], x[(long long)s * DD + t]);
}

// ---------------------------------------------------------------- sum/max(deg,1) (+ optional addend), in place, float4
__global__ void finalize_mean(float* __restrict__ sum, const float* __restrict__ deg,
                              const float* __restrict__ addend, int N) {
  long long n4 = (long long)N * (DD / 4);
  long long i = (long long)blockIdx.x * blockDim.x + threadIdx.x;
  long long stride = (long long)gridDim.x * blockDim.x;
  float4* s4 = reinterpret_cast<float4*>(sum);
  const float4* a4 = reinterpret_cast<const float4*>(addend);
  for (; i < n4; i += stride) {
    int row = (int)(i >> 6);                    // DD/4 == 64 float4 per row
    float r = 1.0f / fmaxf(deg[row], 1.0f);
    float4 v = s4[i];
    v.x *= r; v.y *= r; v.z *= r; v.w *= r;
    if (addend) {
      float4 a = a4[i];
      v.x += a.x; v.y += a.y; v.z += a.z; v.w += a.w;
    }
    s4[i] = v;
  }
}

// ---------------------------------------------------------------- pre-swizzle f32 weights into WMMA B-fragment f16 layout
// Wf[((kt*16+nt)*32 + lane)*16 + j] = W[K][N] with
//   lane<16 : K = kt*32 + j + (j>=8 ? 8 : 0)       N = nt*16 + lane
//   lane>=16: K = kt*32 + 8 + j + (j>=8 ? 8 : 0)   N = nt*16 + lane-16
// kt-slices (16KB) are contiguous -> async-copy friendly.
__global__ void swizzle_w(const float* __restrict__ W, _Float16* __restrict__ Wf) {
  int tid = blockIdx.x * blockDim.x + threadIdx.x;   // 65536 total
  if (tid >= DD * DD) return;
  int j    = tid & 15;
  int lane = (tid >> 4) & 31;
  int rem  = tid >> 9;
  int nt   = rem & 15;
  int kt   = rem >> 4;
  int K  = kt * 32 + ((lane >= 16) ? 8 : 0) + j + ((j >= 8) ? 8 : 0);
  int Nc = nt * 16 + (lane & 15);
  Wf[tid] = (_Float16)W[K * DD + Nc];
}

// issue one 16KB weight slice (kt) into LDS double-buffer `buf` via async global->LDS.
// LDS offset derived from the real shared-memory pointer (low 32 bits of the
// generic address == DS byte offset) so the object escapes into the asm and the
// compiler cannot fold the subsequent LDS reads.
__device__ __forceinline__ void issue_slice(const _Float16* __restrict__ W,
                                            _Float16* sbase,
                                            int kt, int buf, int tid) {
  const char* g = (const char*)(W + (size_t)kt * SLICE_H);
  char* l = (char*)sbase + (size_t)buf * SLICE_B + (size_t)tid * 16;
  #pragma unroll
  for (int r = 0; r < 4; ++r) {
    unsigned long long ga = (unsigned long long)(g + r * 4096 + tid * 16);
    unsigned loff = (unsigned)(unsigned long long)(l + r * 4096);
    asm volatile("global_load_async_to_lds_b128 %0, %1, off"
                 :: "v"(loff), "v"(ga)
                 : "memory");
  }
}

// ---------------------------------------------------------------- WMMA GEMM: out = act(A1@W1 [+ A2@W2] + bias)
// Wave tile = 16 rows x 256 cols (16 f32 accumulators). 8 waves / block.
// B fragments staged in LDS (double-buffered async copy), consumed via ds_load.
__global__ void __launch_bounds__(256)
gemm_wmma(const float* __restrict__ A1, const _Float16* __restrict__ W1,
          const float* __restrict__ A2, const _Float16* __restrict__ W2,
          const float* __restrict__ bias, float* __restrict__ out,
          int Mtiles, int do_relu) {
  __shared__ _Float16 sB[2 * SLICE_H];           // 32 KB double buffer

  const int tid  = threadIdx.x;
  const int wave = tid >> 5;
  const int lane = tid & 31;
  int mtile = blockIdx.x * 8 + wave;
  const bool active = (mtile < Mtiles);
  if (!active) mtile = Mtiles - 1;               // clamp A reads; store is guarded
  const int lm = lane & 15;
  const int hi = lane >> 4;                      // 0 or 1

  v8f zero = {0.f,0.f,0.f,0.f,0.f,0.f,0.f,0.f};
  v8f acc[NT];
  #pragma unroll
  for (int nt = 0; nt < NT; ++nt) acc[nt] = zero;

  const int npasses = A2 ? 2 : 1;
  #pragma unroll 1
  for (int pass = 0; pass < npasses; ++pass) {
    const float*    A = pass ? A2 : A1;
    const _Float16* W = pass ? W2 : W1;
    const float* Arow = A + (long long)(mtile * 16 + lm) * DD;

    issue_slice(W, sB, 0, 0, tid);               // prefetch kt=0
    #pragma unroll 1
    for (int kt = 0; kt < KT; ++kt) {
      if (kt + 1 < KT) {
        issue_slice(W, sB, kt + 1, (kt + 1) & 1, tid); // overlap next slice copy
        asm volatile("s_wait_asynccnt 0x4" ::: "memory");  // current slice landed
      } else {
        asm volatile("s_wait_asynccnt 0x0" ::: "memory");
      }
      __syncthreads();                            // all waves' parts visible

      // A fragment (16x32 f16): lane<16 -> K {0..7,16..23}; lane>=16 -> K {8..15,24..31}
      const float* ap = Arow + kt * 32 + hi * 8;
      float4 a0 = *reinterpret_cast<const float4*>(ap);
      float4 a1 = *reinterpret_cast<const float4*>(ap + 4);
      float4 a2 = *reinterpret_cast<const float4*>(ap + 16);
      float4 a3 = *reinterpret_cast<const float4*>(ap + 20);
      v16h a;
      a[0]=(_Float16)a0.x; a[1]=(_Float16)a0.y; a[2]=(_Float16)a0.z; a[3]=(_Float16)a0.w;
      a[4]=(_Float16)a1.x; a[5]=(_Float16)a1.y; a[6]=(_Float16)a1.z; a[7]=(_Float16)a1.w;
      a[8]=(_Float16)a2.x; a[9]=(_Float16)a2.y; a[10]=(_Float16)a2.z; a[11]=(_Float16)a2.w;
      a[12]=(_Float16)a3.x; a[13]=(_Float16)a3.y; a[14]=(_Float16)a3.z; a[15]=(_Float16)a3.w;

      const _Float16* bbase = &sB[(kt & 1) * SLICE_H + lane * 16];
      #pragma unroll
      for (int nt = 0; nt < NT; ++nt) {
        v16h b = *reinterpret_cast<const v16h*>(bbase + nt * 512);   // ds_load_b128 x2
        acc[nt] = __builtin_amdgcn_wmma_f32_16x16x32_f16(
            false, a, false, b, (short)0, acc[nt], false, false);
      }
      __syncthreads();                            // buffer reuse guard
    }
  }

  if (active) {
    // C layout: VGPR r -> (M = r + 8*hi, N = lane&15)
    #pragma unroll
    for (int nt = 0; nt < NT; ++nt) {
      float bv = bias[nt * 16 + lm];
      #pragma unroll
      for (int r = 0; r < 8; ++r) {
        float v = acc[nt][r] + bv;
        if (do_relu) v = fmaxf(v, 0.0f);
        out[(long long)(mtile * 16 + hi * 8 + r) * DD + nt * 16 + lm] = v;
      }
    }
  }
}

// ---------------------------------------------------------------- host
extern "C" void kernel_launch(void* const* d_in, const int* in_sizes, int n_in,
                              void* d_out, int out_size, void* d_ws, size_t ws_size,
                              hipStream_t stream) {
  (void)n_in; (void)out_size; (void)ws_size;
  const int N = in_sizes[0] / DD;     // 50000
  const int E = in_sizes[1] / DD;     // 500000

  const float* x         = (const float*)d_in[0];
  const float* edge_attr = (const float*)d_in[1];
  const float* W_ne      = (const float*)d_in[2];
  const float* b_ne      = (const float*)d_in[3];
  const float* Wl1       = (const float*)d_in[4];
  const float* bl1       = (const float*)d_in[5];
  const float* Wr1       = (const float*)d_in[6];
  const float* Wl2       = (const float*)d_in[7];
  const float* bl2       = (const float*)d_in[8];
  const float* Wr2       = (const float*)d_in[9];
  const int*   ei        = (const int*)d_in[10];  // JAX default x64-off -> int32; [2,E]
  const int*   src = ei;
  const int*   dst = ei + E;

  // workspace carve-up
  char* ws = (char*)d_ws;
  size_t off = 0;
  auto carve = [&](size_t bytes) -> void* {
    off = (off + 255) & ~(size_t)255;
    void* p = ws + off;
    off += bytes;
    return p;
  };
  float*    sum = (float*)carve((size_t)N * DD * sizeof(float));  // reused accumulator / mean
  float*    x1  = (float*)carve((size_t)N * DD * sizeof(float));
  float*    x2  = (float*)carve((size_t)N * DD * sizeof(float));
  float*    deg = (float*)carve((size_t)N * sizeof(float));
  _Float16* Wf  = (_Float16*)carve((size_t)5 * DD * DD * sizeof(_Float16));
  _Float16 *Wf_ne = Wf,            *Wf_l1 = Wf + DD*DD, *Wf_r1 = Wf + 2*DD*DD,
           *Wf_l2 = Wf + 3*DD*DD,  *Wf_r2 = Wf + 4*DD*DD;

  const int Mtiles = (N + 15) / 16;              // 3125 (N divisible by 16)
  dim3 gblk(256);
  dim3 ggrid((Mtiles + 7) / 8);
  const long long ND = (long long)N * DD;

  // weight swizzle (once per launch; tiny)
  swizzle_w<<<(DD*DD + 255)/256, 256, 0, stream>>>(W_ne, Wf_ne);
  swizzle_w<<<(DD*DD + 255)/256, 256, 0, stream>>>(Wl1,  Wf_l1);
  swizzle_w<<<(DD*DD + 255)/256, 256, 0, stream>>>(Wr1,  Wf_r1);
  swizzle_w<<<(DD*DD + 255)/256, 256, 0, stream>>>(Wl2,  Wf_l2);
  swizzle_w<<<(DD*DD + 255)/256, 256, 0, stream>>>(Wr2,  Wf_r2);

  // degree (shared by all three scatter-means)
  zero_f32<<<256, 256, 0, stream>>>(deg, N);
  deg_count<<<(E + 255)/256, 256, 0, stream>>>(dst, deg, E);

  // 1) e_msg = seg_mean(edge_attr, dst); t0 = x + e_msg  (in sum)
  zero_f32<<<2048, 256, 0, stream>>>(sum, ND);
  scatter_edge<<<E, DD, 0, stream>>>(edge_attr, dst, sum, E);
  finalize_mean<<<2048, 256, 0, stream>>>(sum, deg, x, N);

  // 2) x1 = t0 @ W_ne + b_ne
  gemm_wmma<<<ggrid, gblk, 0, stream>>>(sum, Wf_ne, nullptr, nullptr, b_ne, x1, Mtiles, 0);

  // 3) aggr1 = seg_mean(x1[src], dst); x2 = relu(aggr1@Wl1 + x1@Wr1 + bl1)
  zero_f32<<<2048, 256, 0, stream>>>(sum, ND);
  scatter_gather<<<E, DD, 0, stream>>>(x1, src, dst, sum, E);
  finalize_mean<<<2048, 256, 0, stream>>>(sum, deg, nullptr, N);
  gemm_wmma<<<ggrid, gblk, 0, stream>>>(sum, Wf_l1, x1, Wf_r1, bl1, x2, Mtiles, 1);

  // 4) aggr2 = seg_mean(x2[src], dst); out = aggr2@Wl2 + x2@Wr2 + bl2
  zero_f32<<<2048, 256, 0, stream>>>(sum, ND);
  scatter_gather<<<E, DD, 0, stream>>>(x2, src, dst, sum, E);
  finalize_mean<<<2048, 256, 0, stream>>>(sum, deg, nullptr, N);
  gemm_wmma<<<ggrid, gblk, 0, stream>>>(sum, Wf_l2, x2, Wf_r2, bl2, (float*)d_out, Mtiles, 0);

  // 5) edge_attr passthrough into second output slot
  hipMemcpyAsync((float*)d_out + ND, edge_attr, (size_t)E * DD * sizeof(float),
                 hipMemcpyDeviceToDevice, stream);
}
